// NonLocalBlock_46110768890082
// MI455X (gfx1250) — matrix operand order repaired and checked by reading, not compiled, not run on hardware
//
#include <hip/hip_runtime.h>
#include <hip/hip_bf16.h>

// ---------------------------------------------------------------------------
// NonLocalBlock (SAGAN self-attention) for MI455X / gfx1250.
// All GEMMs run on v_wmma_f32_16x16x32_bf16; attention is flash-style
// (online softmax, no materialized [B,4096,1024] score tensor).
// ---------------------------------------------------------------------------

#define Bn   8
#define CIN  256
#define HW   4096     // 64*64
#define HWP  1024     // 32*32 after 2x2 maxpool

typedef __attribute__((ext_vector_type(16))) __bf16 v16bf;
typedef __attribute__((ext_vector_type(8)))  float  v8f;

union AB { v16bf v; uint4 q[2]; };

#define WMMA_BF16(A, B, C) \
  __builtin_amdgcn_wmma_f32_16x16x32_bf16(false, (A), false, (B), (short)0, (C), false, false)

static __device__ __forceinline__ v8f vzero() {
  v8f v;
#pragma unroll
  for (int i = 0; i < 8; ++i) v[i] = 0.0f;
  return v;
}

static __device__ __forceinline__ unsigned short f2bf(float f) {
  unsigned u = __float_as_uint(f);
  unsigned r = u + 0x7FFFu + ((u >> 16) & 1u);   // round-to-nearest-even
  return (unsigned short)(r >> 16);
}

static __device__ __forceinline__ float bf2f(unsigned short u) {
  return __uint_as_float(((unsigned)u) << 16);
}

// ---------------------------------------------------------------------------
// K0: x [B,256,4096] f32  ->  xT [B,4096,256] bf16   (LDS tile transpose)
// ---------------------------------------------------------------------------
__global__ void transpose_x(const float* __restrict__ x, unsigned short* __restrict__ xT) {
  __shared__ float t[32][33];
  const int b = blockIdx.z, n0 = blockIdx.x * 32, c0 = blockIdx.y * 32;
  const int tx = threadIdx.x, ty = threadIdx.y;
  for (int j = ty; j < 32; j += 8)
    t[j][tx] = x[((size_t)b * CIN + c0 + j) * HW + n0 + tx];
  __syncthreads();
  for (int j = ty; j < 32; j += 8)
    xT[((size_t)b * HW + n0 + j) * CIN + c0 + tx] = f2bf(t[tx][j]);
}

// ---------------------------------------------------------------------------
// K0b: weights f32 -> bf16, equalized-LR scale sqrt(2/256) folded in
// ---------------------------------------------------------------------------
__global__ void convert_w(const float* __restrict__ w, unsigned short* __restrict__ o, int n) {
  int i = blockIdx.x * 256 + threadIdx.x;
  if (i < n) o[i] = f2bf(w[i] * 0.0883883476483184405f);
}

// ---------------------------------------------------------------------------
// K1: Y[b][n][co] = sum_ci xT[b][n][ci] * W[co][ci]   (one wave = 16x16 tile)
// A = xT tile (row-major, K contiguous), B = W (co rows, ci contiguous).
// ---------------------------------------------------------------------------
__global__ __launch_bounds__(32) void proj_gemm(const unsigned short* __restrict__ xT,
                                                const unsigned short* __restrict__ W,
                                                unsigned short* __restrict__ Y, int Cco) {
  const int lane = threadIdx.x, half = lane >> 4, l15 = lane & 15;
  const int n0 = blockIdx.x << 4, c0 = blockIdx.y << 4, b = blockIdx.z;
  const unsigned short* xrow = xT + ((size_t)b * HW + n0 + l15) * CIN;
  const unsigned short* wrow = W + (size_t)(c0 + l15) * CIN + half * 16;
  v8f acc = vzero();
#pragma unroll
  for (int k = 0; k < CIN; k += 32) {
    AB a, bb;
    a.q[0]  = *(const uint4*)(xrow + k + half * 16);
    a.q[1]  = *(const uint4*)(xrow + k + half * 16 + 8);
    bb.q[0] = *(const uint4*)(wrow + k);
    bb.q[1] = *(const uint4*)(wrow + k + 8);
    acc = WMMA_BF16(a.v, bb.v, acc);
  }
  unsigned short* yb = Y + (size_t)b * HW * Cco;
#pragma unroll
  for (int i = 0; i < 8; ++i)
    yb[(size_t)(n0 + i + 8 * half) * Cco + c0 + l15] = f2bf(acc[i]);
}

// ---------------------------------------------------------------------------
// K2a: 2x2 maxpool, phiFullT [B,4096,64] -> phiT [B,1024,64] (m-major, c contig)
// ---------------------------------------------------------------------------
__global__ void pool_phi(const unsigned short* __restrict__ pf, unsigned short* __restrict__ pt) {
  int idx = blockIdx.x * 256 + threadIdx.x;           // B*1024*64 threads
  int c = idx & 63, m = (idx >> 6) & 1023, b = idx >> 16;
  int hp = m >> 5, wp = m & 31;
  int n00 = hp * 128 + wp * 2;
  const unsigned short* base = pf + (size_t)b * HW * 64;
  float v0 = bf2f(base[(size_t)(n00     ) * 64 + c]);
  float v1 = bf2f(base[(size_t)(n00 +  1) * 64 + c]);
  float v2 = bf2f(base[(size_t)(n00 + 64) * 64 + c]);
  float v3 = bf2f(base[(size_t)(n00 + 65) * 64 + c]);
  pt[idx] = f2bf(fmaxf(fmaxf(v0, v1), fmaxf(v2, v3)));
}

// ---------------------------------------------------------------------------
// K2b: 2x2 maxpool + transpose, gFullT [B,4096,256] -> g [B,256,1024] (c-major)
// ---------------------------------------------------------------------------
__global__ void pool_g(const unsigned short* __restrict__ gf, unsigned short* __restrict__ g) {
  int idx = blockIdx.x * 256 + threadIdx.x;           // B*256*1024 threads
  int m = idx & 1023, c = (idx >> 10) & 255, b = idx >> 18;
  int hp = m >> 5, wp = m & 31;
  int n00 = hp * 128 + wp * 2;
  const unsigned short* base = gf + (size_t)b * HW * CIN;
  float v0 = bf2f(base[(size_t)(n00     ) * CIN + c]);
  float v1 = bf2f(base[(size_t)(n00 +  1) * CIN + c]);
  float v2 = bf2f(base[(size_t)(n00 + 64) * CIN + c]);
  float v3 = bf2f(base[(size_t)(n00 + 65) * CIN + c]);
  g[idx] = f2bf(fmaxf(fmaxf(v0, v1), fmaxf(v2, v3)));
}

// ---------------------------------------------------------------------------
// K3: flash attention. One wave handles a 16-query tile against all 1024 keys.
//   pass 1: online row max / sum of exp (scores via WMMA, stats via shfl)
//   pass 2: recompute scores, P = exp(S - max) relayout through LDS,
//           att[16,256] accumulated via WMMA, scaled by 1/sum at the end.
// Output attT [B,4096,256] bf16 (n-major, c contiguous).
// ---------------------------------------------------------------------------
__global__ __launch_bounds__(32) void attn_kernel(const unsigned short* __restrict__ thetaT,
                                                  const unsigned short* __restrict__ phiT,
                                                  const unsigned short* __restrict__ gP,
                                                  unsigned short* __restrict__ attT) {
  __shared__ alignas(16) unsigned short Pl[16 * 32];  // P tile, row-major [n][m]
  const int lane = threadIdx.x, half = lane >> 4, l15 = lane & 15;
  const int b = blockIdx.y;
  const int n0 = blockIdx.x << 4;

  // theta A tiles (K = channel dim, 64 -> two K=32 tiles), resident all loop
  const unsigned short* thRow = thetaT + ((size_t)b * HW + n0 + l15) * 64;
  AB a0, a1;
  a0.q[0] = *(const uint4*)(thRow + half * 16);
  a0.q[1] = *(const uint4*)(thRow + 8 + half * 16);
  a1.q[0] = *(const uint4*)(thRow + 32 + half * 16);
  a1.q[1] = *(const uint4*)(thRow + 40 + half * 16);

  const unsigned short* phiBase = phiT + (size_t)b * HWP * 64;

  float rmax[8], rsum[8];
#pragma unroll
  for (int i = 0; i < 8; ++i) { rmax[i] = -3.0e38f; rsum[i] = 0.0f; }

  // ---- pass 1: online softmax statistics ---------------------------------
  for (int m0 = 0; m0 < HWP; m0 += 16) {
    const unsigned short* pr = phiBase + (size_t)(m0 + l15) * 64 + half * 16;
    AB b0, b1;
    b0.q[0] = *(const uint4*)(pr);
    b0.q[1] = *(const uint4*)(pr + 8);
    b1.q[0] = *(const uint4*)(pr + 32);
    b1.q[1] = *(const uint4*)(pr + 40);
    v8f s = vzero();
    s = WMMA_BF16(a0.v, b0.v, s);
    s = WMMA_BF16(a1.v, b1.v, s);
#pragma unroll
    for (int i = 0; i < 8; ++i) {
      float v = s[i];
      float tm = v;
#pragma unroll
      for (int off = 8; off >= 1; off >>= 1) tm = fmaxf(tm, __shfl_xor(tm, off, 16));
      float nm = fmaxf(rmax[i], tm);
      float e = __expf(v - nm);
#pragma unroll
      for (int off = 8; off >= 1; off >>= 1) e += __shfl_xor(e, off, 16);
      rsum[i] = rsum[i] * __expf(rmax[i] - nm) + e;
      rmax[i] = nm;
    }
  }

  // ---- pass 2: P x g accumulation ----------------------------------------
  v8f acc[16];
#pragma unroll
  for (int t = 0; t < 16; ++t) acc[t] = vzero();
  const unsigned short* gBase = gP + (size_t)b * CIN * HWP;

  for (int m0 = 0; m0 < HWP; m0 += 32) {
#pragma unroll
    for (int sub = 0; sub < 2; ++sub) {
      const int mm = m0 + sub * 16;
      const unsigned short* pr = phiBase + (size_t)(mm + l15) * 64 + half * 16;
      AB b0, b1;
      b0.q[0] = *(const uint4*)(pr);
      b0.q[1] = *(const uint4*)(pr + 8);
      b1.q[0] = *(const uint4*)(pr + 32);
      b1.q[1] = *(const uint4*)(pr + 40);
      v8f s = vzero();
      s = WMMA_BF16(a0.v, b0.v, s);
      s = WMMA_BF16(a1.v, b1.v, s);
#pragma unroll
      for (int i = 0; i < 8; ++i) {
        float p = __expf(s[i] - rmax[i]);
        Pl[(i + 8 * half) * 32 + sub * 16 + l15] = f2bf(p);   // C-layout -> row-major
      }
    }
    __syncthreads();
    AB pa;                                                   // reload as A-layout
    const unsigned short* prow = &Pl[l15 * 32];
    pa.q[0] = *(const uint4*)(prow + half * 16);
    pa.q[1] = *(const uint4*)(prow + 8 + half * 16);
    __syncthreads();
#pragma unroll
    for (int t = 0; t < 16; ++t) {
      const unsigned short* gr = gBase + (size_t)(t * 16 + l15) * HWP + m0 + half * 16;
      AB gb;
      gb.q[0] = *(const uint4*)(gr);
      gb.q[1] = *(const uint4*)(gr + 8);
      acc[t] = WMMA_BF16(pa.v, gb.v, acc[t]);
    }
  }

  // ---- epilogue: divide by softmax sum, store attT bf16 ------------------
  float rinv[8];
#pragma unroll
  for (int i = 0; i < 8; ++i) rinv[i] = 1.0f / rsum[i];
  unsigned short* ob = attT + (size_t)b * HW * CIN;
#pragma unroll
  for (int t = 0; t < 16; ++t)
#pragma unroll
    for (int i = 0; i < 8; ++i)
      ob[(size_t)(n0 + i + 8 * half) * CIN + t * 16 + l15] = f2bf(acc[t][i] * rinv[i]);
}

// ---------------------------------------------------------------------------
// K4: out[b][co][n] = (gamma * (attT x w_o^T) + (xT x w_res^T)) / sqrt(2)
// ---------------------------------------------------------------------------
__global__ __launch_bounds__(32) void out_kernel(const unsigned short* __restrict__ attT,
                                                 const unsigned short* __restrict__ xT,
                                                 const unsigned short* __restrict__ wo,
                                                 const unsigned short* __restrict__ wr,
                                                 const float* __restrict__ gamma,
                                                 float* __restrict__ out) {
  const int lane = threadIdx.x, half = lane >> 4, l15 = lane & 15;
  const int n0 = blockIdx.x << 4, c0 = blockIdx.y << 4, b = blockIdx.z;
  const unsigned short* arow  = attT + ((size_t)b * HW + n0 + l15) * CIN;
  const unsigned short* xrow  = xT   + ((size_t)b * HW + n0 + l15) * CIN;
  const unsigned short* worow = wo + (size_t)(c0 + l15) * CIN + half * 16;
  const unsigned short* wrrow = wr + (size_t)(c0 + l15) * CIN + half * 16;
  v8f ao = vzero(), ar = vzero();
#pragma unroll
  for (int k = 0; k < CIN; k += 32) {
    AB a, x, bo, br;
    a.q[0]  = *(const uint4*)(arow + k + half * 16);
    a.q[1]  = *(const uint4*)(arow + k + half * 16 + 8);
    x.q[0]  = *(const uint4*)(xrow + k + half * 16);
    x.q[1]  = *(const uint4*)(xrow + k + half * 16 + 8);
    bo.q[0] = *(const uint4*)(worow + k);
    bo.q[1] = *(const uint4*)(worow + k + 8);
    br.q[0] = *(const uint4*)(wrrow + k);
    br.q[1] = *(const uint4*)(wrrow + k + 8);
    ao = WMMA_BF16(a.v, bo.v, ao);
    ar = WMMA_BF16(x.v, br.v, ar);
  }
  const float gm = gamma[0];
  float* ob = out + (size_t)b * 512 * HW + (size_t)(c0 + l15) * HW + n0;
#pragma unroll
  for (int i = 0; i < 8; ++i)
    ob[i + 8 * half] = (gm * ao[i] + ar[i]) * 0.70710678118654752f;
}

// ---------------------------------------------------------------------------
extern "C" void kernel_launch(void* const* d_in, const int* in_sizes, int n_in,
                              void* d_out, int out_size, void* d_ws, size_t ws_size,
                              hipStream_t stream) {
  (void)in_sizes; (void)n_in; (void)out_size; (void)ws_size;
  const float* x     = (const float*)d_in[0];
  const float* wth   = (const float*)d_in[1];
  const float* wph   = (const float*)d_in[2];
  const float* wg    = (const float*)d_in[3];
  const float* wo    = (const float*)d_in[4];
  const float* wres  = (const float*)d_in[5];
  const float* gamma = (const float*)d_in[6];
  float* out = (float*)d_out;

  char* ws = (char*)d_ws;
  size_t off = 0;
  auto carve = [&](size_t bytes) {
    char* p = ws + off;
    off += (bytes + 255) & ~(size_t)255;
    return p;
  };
  unsigned short* xT      = (unsigned short*)carve((size_t)Bn * HW * CIN * 2);   // 16 MB
  unsigned short* thetaT  = (unsigned short*)carve((size_t)Bn * HW * 64 * 2);    //  4 MB
  unsigned short* phiFull = (unsigned short*)carve((size_t)Bn * HW * 64 * 2);    //  4 MB
  unsigned short* phiTl   = (unsigned short*)carve((size_t)Bn * HWP * 64 * 2);   //  1 MB
  unsigned short* gFull   = (unsigned short*)carve((size_t)Bn * HW * CIN * 2);   // 16 MB
  unsigned short* gPool   = (unsigned short*)carve((size_t)Bn * CIN * HWP * 2);  //  4 MB
  unsigned short* attT    = (unsigned short*)carve((size_t)Bn * HW * CIN * 2);   // 16 MB
  unsigned short* wthB    = (unsigned short*)carve((size_t)64 * CIN * 2);
  unsigned short* wphB    = (unsigned short*)carve((size_t)64 * CIN * 2);
  unsigned short* wgB     = (unsigned short*)carve((size_t)CIN * CIN * 2);
  unsigned short* woB     = (unsigned short*)carve((size_t)512 * CIN * 2);
  unsigned short* wresB   = (unsigned short*)carve((size_t)512 * CIN * 2);

  // K0: transpose + downconvert x
  transpose_x<<<dim3(HW / 32, CIN / 32, Bn), dim3(32, 8), 0, stream>>>(x, xT);

  // K0b: weights -> bf16 with equalized-LR scale
  convert_w<<<(64 * CIN + 255) / 256, 256, 0, stream>>>(wth,  wthB,  64 * CIN);
  convert_w<<<(64 * CIN + 255) / 256, 256, 0, stream>>>(wph,  wphB,  64 * CIN);
  convert_w<<<(CIN * CIN + 255) / 256, 256, 0, stream>>>(wg,  wgB,  CIN * CIN);
  convert_w<<<(512 * CIN + 255) / 256, 256, 0, stream>>>(wo,  woB,  512 * CIN);
  convert_w<<<(512 * CIN + 255) / 256, 256, 0, stream>>>(wres, wresB, 512 * CIN);

  // K1: projection GEMMs (WMMA)
  proj_gemm<<<dim3(HW / 16,  4, Bn), 32, 0, stream>>>(xT, wthB, thetaT, 64);
  proj_gemm<<<dim3(HW / 16,  4, Bn), 32, 0, stream>>>(xT, wphB, phiFull, 64);
  proj_gemm<<<dim3(HW / 16, 16, Bn), 32, 0, stream>>>(xT, wgB,  gFull, 256);

  // K2: 2x2 maxpool of phi and g
  pool_phi<<<(Bn * HWP * 64) / 256, 256, 0, stream>>>(phiFull, phiTl);
  pool_g<<<(Bn * CIN * HWP) / 256, 256, 0, stream>>>(gFull, gPool);

  // K3: flash attention (WMMA scores + WMMA P*g)
  attn_kernel<<<dim3(HW / 16, Bn), 32, 0, stream>>>(thetaT, phiTl, gPool, attT);

  // K4: fused output + residual + gamma/invsqrt2 epilogue (WMMA)
  out_kernel<<<dim3(HW / 16, 512 / 16, Bn), 32, 0, stream>>>(attT, xT, woB, wresB, gamma, out);
}
